// TrainableVarRegBlock_44014824849813
// MI455X (gfx1250) — matrix operand order repaired and checked by reading, not compiled, not run on hardware
//
#include <hip/hip_runtime.h>

typedef __attribute__((ext_vector_type(16))) _Float16 v16h;
typedef __attribute__((ext_vector_type(8)))  _Float16 v8h;
typedef __attribute__((ext_vector_type(8)))  float    v8f;
typedef __attribute__((ext_vector_type(4)))  float    v4f;

// ----------------------------------------------------------------------------
// helpers
// ----------------------------------------------------------------------------
__device__ __forceinline__ float mish_f(float x) {
    float sp = (x > 20.f) ? x : log1pf(expf(x));
    return x * tanhf(sp);
}

__global__ void fill_zero_kernel(float* p, int n) {
    int i = blockIdx.x * blockDim.x + threadIdx.x;
    if (i < n) p[i] = 0.f;
}

__global__ void scale_copy_kernel(const float* __restrict__ in, float* __restrict__ out,
                                  float s, int n) {
    int i = blockIdx.x * blockDim.x + threadIdx.x;
    if (i < n) out[i] = in[i] * s;
}

// ----------------------------------------------------------------------------
// trilinear resize (jax.image.resize 'trilinear': half-pixel centers, clamp)
// ----------------------------------------------------------------------------
__global__ void resize_tri_kernel(const float* __restrict__ in, float* __restrict__ out,
                                  int C, int Di, int Hi, int Wi,
                                  int Do, int Ho, int Wo, float mul) {
    int i = blockIdx.x * blockDim.x + threadIdx.x;
    int No = Do * Ho * Wo;
    if (i >= C * No) return;
    int c = i / No;
    int v = i - c * No;
    int xo = v % Wo;
    int yo = (v / Wo) % Ho;
    int zo = v / (Wo * Ho);
    float sz = (zo + 0.5f) * (float)Di / (float)Do - 0.5f;
    float sy = (yo + 0.5f) * (float)Hi / (float)Ho - 0.5f;
    float sx = (xo + 0.5f) * (float)Wi / (float)Wo - 0.5f;
    sz = fminf(fmaxf(sz, 0.f), (float)(Di - 1));
    sy = fminf(fmaxf(sy, 0.f), (float)(Hi - 1));
    sx = fminf(fmaxf(sx, 0.f), (float)(Wi - 1));
    int z0 = (int)sz, y0 = (int)sy, x0 = (int)sx;
    float fz = sz - z0, fy = sy - y0, fx = sx - x0;
    int z1 = min(z0 + 1, Di - 1), y1 = min(y0 + 1, Hi - 1), x1 = min(x0 + 1, Wi - 1);
    const float* b = in + (size_t)c * Di * Hi * Wi;
    auto g = [&](int z, int y, int x) { return b[((size_t)z * Hi + y) * Wi + x]; };
    float c00 = g(z0, y0, x0) * (1.f - fx) + g(z0, y0, x1) * fx;
    float c01 = g(z0, y1, x0) * (1.f - fx) + g(z0, y1, x1) * fx;
    float c10 = g(z1, y0, x0) * (1.f - fx) + g(z1, y0, x1) * fx;
    float c11 = g(z1, y1, x0) * (1.f - fx) + g(z1, y1, x1) * fx;
    float c0 = c00 * (1.f - fy) + c01 * fy;
    float c1 = c10 * (1.f - fy) + c11 * fy;
    out[i] = mul * (c0 * (1.f - fz) + c1 * fz);
}

// ----------------------------------------------------------------------------
// trilinear warp with border clamp (single-channel image, 3-channel flow z,y,x)
// ----------------------------------------------------------------------------
__global__ void warp_kernel(const float* __restrict__ img, const float* __restrict__ vf,
                            float* __restrict__ out, int D, int H, int W) {
    int idx = blockIdx.x * blockDim.x + threadIdx.x;
    int N = D * H * W;
    if (idx >= N) return;
    int w = idx % W;
    int h = (idx / W) % H;
    int d = idx / (W * H);
    float z = fminf(fmaxf((float)d + vf[idx],         0.f), (float)(D - 1));
    float y = fminf(fmaxf((float)h + vf[N + idx],     0.f), (float)(H - 1));
    float x = fminf(fmaxf((float)w + vf[2 * N + idx], 0.f), (float)(W - 1));
    int z0 = (int)z, y0 = (int)y, x0 = (int)x;
    float fz = z - z0, fy = y - y0, fx = x - x0;
    int z1 = min(z0 + 1, D - 1), y1 = min(y0 + 1, H - 1), x1 = min(x0 + 1, W - 1);
    auto g = [&](int zz, int yy, int xx) { return img[((size_t)zz * H + yy) * W + xx]; };
    float c00 = g(z0, y0, x0) * (1.f - fx) + g(z0, y0, x1) * fx;
    float c01 = g(z0, y1, x0) * (1.f - fx) + g(z0, y1, x1) * fx;
    float c10 = g(z1, y0, x0) * (1.f - fx) + g(z1, y0, x1) * fx;
    float c11 = g(z1, y1, x0) * (1.f - fx) + g(z1, y1, x1) * fx;
    float c0 = c00 * (1.f - fy) + c01 * fy;
    float c1 = c10 * (1.f - fy) + c11 * fy;
    out[idx] = c0 * (1.f - fz) + c1 * fz;
}

// ----------------------------------------------------------------------------
// demon forces + feature map, written into halo-padded NHWC f16 buffer with
// 16-channel rows: [fz, fy, fx, fixed, warped, 0..0] -> two b128 stores/voxel
// ----------------------------------------------------------------------------
__global__ void feat_kernel(const float* __restrict__ warped, const float* __restrict__ fixed,
                            _Float16* __restrict__ feat, int D, int H, int W) {
    int idx = blockIdx.x * blockDim.x + threadIdx.x;
    int N = D * H * W;
    if (idx >= N) return;
    int w = idx % W;
    int h = (idx / W) % H;
    int d = idx / (W * H);
    float wc = warped[idx];
    float fi = fixed[idx];
    float diff = wc - fi;
    auto at = [&](int z, int y, int x) { return warped[((size_t)z * H + y) * W + x]; };
    // jnp.gradient: one-sided at edges, central in interior (spacing 1)
    float gz = (d == 0)     ? at(1, h, w) - wc
             : (d == D - 1) ? wc - at(D - 2, h, w)
                            : 0.5f * (at(d + 1, h, w) - at(d - 1, h, w));
    float gy = (h == 0)     ? at(d, 1, w) - wc
             : (h == H - 1) ? wc - at(d, H - 2, w)
                            : 0.5f * (at(d, h + 1, w) - at(d, h - 1, w));
    float gx = (w == 0)     ? at(d, h, 1) - wc
             : (w == W - 1) ? wc - at(d, h, W - 2)
                            : 0.5f * (at(d, h, w + 1) - at(d, h, w - 1));
    float denom = gz * gz + gy * gy + gx * gx + diff * diff + 1e-9f;
    float s = diff / denom;
    v8h f = {};
    f[0] = (_Float16)(gz * s);
    f[1] = (_Float16)(gy * s);
    f[2] = (_Float16)(gx * s);
    f[3] = (_Float16)fi;
    f[4] = (_Float16)wc;
    v8h zz = {};
    int Hp = H + 2, Wp = W + 2;
    size_t pidx = ((size_t)(d + 1) * Hp + (h + 1)) * Wp + (w + 1);
    *(v8h*)(feat + pidx * 16) = f;
    *(v8h*)(feat + pidx * 16 + 8) = zz;
}

// ----------------------------------------------------------------------------
// weight pre-pack into exact wave32 B-fragment order for v_wmma_f32_16x16x32_f16
// B layout: N=lane&15, halves j hold K=kt*32+(j>>3)*16+(lane>>4)*8+(j&7)
// k-order: k = tap*CPAD + cin, tap=(dz*3+dy)*3+dx  (matches NHWC A gather).
// Zero for tap>=27 / cin>=Cin / n>=Cout.
// ----------------------------------------------------------------------------
__global__ void pack_weights_kernel(const float* __restrict__ w, _Float16* __restrict__ wpack,
                                    int Cin, int CPAD, int Cout, int Ktiles, int Ntiles) {
    int i = blockIdx.x * blockDim.x + threadIdx.x;
    int total = Ktiles * Ntiles * 512;
    if (i >= total) return;
    int j = i & 15;
    int lane = (i >> 4) & 31;
    int nt = (i >> 9) % Ntiles;
    int kt = i / (512 * Ntiles);
    int row = lane & 15, hi = lane >> 4;
    int n = nt * 16 + row;
    int k = kt * 32 + (j >> 3) * 16 + hi * 8 + (j & 7);
    int tap = k / CPAD;
    int cin = k - tap * CPAD;
    float v = 0.f;
    if (n < Cout && tap < 27 && cin < Cin)
        v = w[((size_t)n * Cin + cin) * 27 + tap];
    wpack[i] = (_Float16)v;
}

// ----------------------------------------------------------------------------
// implicit-GEMM 3x3x3 SAME conv via v_wmma_f32_16x16x32_f16.
// Input: halo-padded (DIM+2)^3 NHWC f16 (halo = zeros -> SAME padding comes
// from memory; every tap load is unconditional). DIM, CPAD (>=16), KTILES
// compile-time: tap is hi-independent, so every A load is a b128 at a pure
// compile-time immediate offset off one per-lane base pointer.
// One wave per 16-voxel M tile; NT 16-wide Cout tiles accumulated per wave.
// K-pad taps (>=27) read own-voxel data; their B weights are 0.
// ----------------------------------------------------------------------------
template <int NT, int CPAD, int KTILES, int DIM>
__global__ void conv3d_wmma_kernel(const _Float16* __restrict__ act,
                                   const _Float16* __restrict__ wpack,
                                   float* __restrict__ out, int Cout) {
    const int DHW = DIM * DIM * DIM;
    const int DP = DIM + 2;
    const int lane = threadIdx.x & 31;
    const int tileM = blockIdx.x * 4 + (threadIdx.x >> 5);
    const int row = lane & 15;
    const int hi = lane >> 4;
    const int m = tileM * 16 + row;
    const int w = m % DIM;
    const int h = (m / DIM) % DIM;
    const int d = m / (DIM * DIM);

    // per-lane A base: padded voxel row start (+ hi*8 channel sub-offset)
    const size_t pbase = ((size_t)(d + 1) * DP + (h + 1)) * DP + (w + 1);
    const _Float16* actm = act + pbase * CPAD + hi * 8;

    v8f acc[NT] = {};

#pragma unroll
    for (int kt = 0; kt < KTILES; ++kt) {
        v16h a;
#pragma unroll
        for (int g = 0; g < 2; ++g) {
            const int kb = kt * 32 + g * 16;              // compile-time
            int tap = kb / CPAD;                          // hi-independent (CPAD>=16)
            if (tap > 26) tap = 13;                       // K-pad: B weight is zero ->
                                                          // load own voxel, contributes 0
            const int dz = tap / 9 - 1, ry = tap % 9;
            const int dy = ry / 3 - 1, dx = ry % 3 - 1;
            const int off = ((dz * DP + dy) * DP + dx) * CPAD + (kb & (CPAD - 1));
            v8h val = *(const v8h*)(actm + off);          // unconditional, halo-safe
#pragma unroll
            for (int q = 0; q < 8; ++q) a[g * 8 + q] = val[q];
        }
#pragma unroll
        for (int nt = 0; nt < NT; ++nt) {
            const v16h b = *(const v16h*)(wpack + (((size_t)kt * NT + nt) * 32 + lane) * 16);
            acc[nt] = __builtin_amdgcn_wmma_f32_16x16x32_f16(
                false, a, false, b, (short)0, acc[nt], false, false);
        }
    }

    // C/D layout: VGPR r -> M = r + hi*8, N = row  => 8 consecutive voxels/lane
    const int mbase = tileM * 16 + hi * 8;
#pragma unroll
    for (int nt = 0; nt < NT; ++nt) {
        int n = nt * 16 + row;
        if (n < Cout) {
            v4f lo = {acc[nt][0], acc[nt][1], acc[nt][2], acc[nt][3]};
            v4f hv = {acc[nt][4], acc[nt][5], acc[nt][6], acc[nt][7]};
            *(v4f*)(out + (size_t)n * DHW + mbase) = lo;
            *(v4f*)(out + (size_t)n * DHW + mbase + 4) = hv;
        }
    }
}

// ----------------------------------------------------------------------------
// instance-norm stats (B=1), conv output is NCHW f32: one block per channel
// ----------------------------------------------------------------------------
__global__ void inorm_stats_kernel(const float* __restrict__ x, float* __restrict__ stats,
                                   int Nvox) {
    int c = blockIdx.x;
    const float* xc = x + (size_t)c * Nvox;
    float s = 0.f, s2 = 0.f;
    for (int i = threadIdx.x; i < Nvox; i += blockDim.x) {
        float v = xc[i];
        s += v;
        s2 += v * v;
    }
    __shared__ float ss[256];
    __shared__ float sq[256];
    int t = threadIdx.x;
    ss[t] = s;
    sq[t] = s2;
    __syncthreads();
    for (int o = 128; o > 0; o >>= 1) {
        if (t < o) { ss[t] += ss[t + o]; sq[t] += sq[t + o]; }
        __syncthreads();
    }
    if (t == 0) {
        float mean = ss[0] / (float)Nvox;
        stats[2 * c] = mean;
        stats[2 * c + 1] = sq[0] / (float)Nvox - mean * mean;  // population var
    }
}

// ----------------------------------------------------------------------------
// inorm + mish, NCHW f32 -> halo-padded NHWC f16 (coalesced channel-plane
// reads, contiguous per-thread 2*C-byte vector writes into interior)
// ----------------------------------------------------------------------------
template <int C>
__global__ void norm_mish_nhwc_kernel(const float* __restrict__ x,
                                      const float* __restrict__ stats,
                                      _Float16* __restrict__ out, int D, int H, int W) {
    int v = blockIdx.x * blockDim.x + threadIdx.x;
    int Nvox = D * H * W;
    if (v >= Nvox) return;
    int w = v % W;
    int h = (v / W) % H;
    int d = v / (W * H);
    int Hp = H + 2, Wp = W + 2;
    size_t pidx = ((size_t)(d + 1) * Hp + (h + 1)) * Wp + (w + 1);
#pragma unroll
    for (int q = 0; q < C / 8; ++q) {
        v8h o;
#pragma unroll
        for (int e = 0; e < 8; ++e) {
            int c = q * 8 + e;
            float mean = stats[2 * c];
            float var = stats[2 * c + 1];
            float y = (x[(size_t)c * Nvox + v] - mean) * rsqrtf(var + 1e-5f);
            o[e] = (_Float16)mish_f(y);
        }
        *(v8h*)(out + pidx * C + q * 8) = o;
    }
}

// ----------------------------------------------------------------------------
// vf += tau * (dv + b4)   (tau = 1)
// ----------------------------------------------------------------------------
__global__ void vf_update_kernel(float* __restrict__ vf, const float* __restrict__ dv,
                                 const float* __restrict__ b4, int Nvox) {
    int idx = blockIdx.x * blockDim.x + threadIdx.x;
    if (idx >= 3 * Nvox) return;
    int c = idx / Nvox;
    vf[idx] += dv[idx] + b4[c];
}

// ----------------------------------------------------------------------------
// zero-padded separable 7-tap Gaussian (sigma=1, cutoff 3), one axis per pass
// ----------------------------------------------------------------------------
__global__ void smooth_axis_kernel(const float* __restrict__ in, float* __restrict__ out,
                                   int C, int D, int H, int W, int axis) {
    int N = D * H * W;
    int idx = blockIdx.x * blockDim.x + threadIdx.x;
    if (idx >= C * N) return;
    int c = idx / N;
    int v = idx - c * N;
    int w = v % W;
    int h = (v / W) % H;
    int d = v / (W * H);
    const float GW[7] = {0.004433048f, 0.054005583f, 0.242036229f, 0.399050281f,
                         0.242036229f, 0.054005583f, 0.004433048f};
    const float* b = in + (size_t)c * N;
    float acc = 0.f;
#pragma unroll
    for (int t = -3; t <= 3; ++t) {
        int z = d, y = h, x = w;
        if (axis == 0) z += t;
        else if (axis == 1) y += t;
        else x += t;
        if ((unsigned)z < (unsigned)D && (unsigned)y < (unsigned)H && (unsigned)x < (unsigned)W)
            acc += GW[t + 3] * b[((size_t)z * H + y) * W + x];
    }
    out[idx] = acc;
}

// ----------------------------------------------------------------------------
// orchestration
// ----------------------------------------------------------------------------
extern "C" void kernel_launch(void* const* d_in, const int* in_sizes, int n_in,
                              void* d_out, int out_size, void* d_ws, size_t ws_size,
                              hipStream_t stream) {
    (void)in_sizes; (void)n_in; (void)out_size; (void)ws_size;
    const float* image  = (const float*)d_in[0];
    // d_in[1] = mask (unused), d_in[3] = spacing (unused)
    const float* moving = (const float*)d_in[2];
    const float* w1 = (const float*)d_in[4];
    const float* w2 = (const float*)d_in[5];
    const float* w3 = (const float*)d_in[6];
    const float* w4 = (const float*)d_in[7];
    const float* b4 = (const float*)d_in[8];

    const int DF = 96;
    const int NF = DF * DF * DF;             // 884736
    const size_t NPF = (size_t)98 * 98 * 98; // padded full-res voxels

    // workspace carve (~475 MB)
    char* p = (char*)d_ws;
    auto carve = [&](size_t bytes) {
        void* r = (void*)p;
        p += (bytes + 255) & ~(size_t)255;
        return r;
    };
    float*    vf     = (float*)carve((size_t)3 * NF * 4);
    float*    vf2    = (float*)carve((size_t)3 * NF * 4);
    float*    si     = (float*)carve((size_t)NF * 4);
    float*    smov   = (float*)carve((size_t)NF * 4);
    float*    warped = (float*)carve((size_t)NF * 4);
    float*    conv   = (float*)carve((size_t)64 * NF * 4);   // NCHW conv outputs
    _Float16* act16  = (_Float16*)carve(NPF * 16 * 2);       // halo-padded NHWC f16
    _Float16* act32  = (_Float16*)carve(NPF * 32 * 2);
    _Float16* act64  = (_Float16*)carve(NPF * 64 * 2);
    _Float16* wpack  = (_Float16*)carve((size_t)1 << 19);
    float*    stats  = (float*)carve(4096);

    _Float16* wp1 = wpack;             // Ktiles=14 (CPAD=16), Ntiles=2 -> 14336 halves
    _Float16* wp2 = wpack + 14336;     // Ktiles=27 (CPAD=32), Ntiles=4 -> 55296
    _Float16* wp3 = wpack + 69632;     // Ktiles=54 (CPAD=64), Ntiles=2 -> 55296
    _Float16* wp4 = wpack + 124928;    // Ktiles=27 (CPAD=32), Ntiles=1 -> 13824

    auto nb = [](int n) { return (n + 255) / 256; };

    pack_weights_kernel<<<nb(14 * 2 * 512), 256, 0, stream>>>(w1, wp1, 5, 16, 32, 14, 2);
    pack_weights_kernel<<<nb(27 * 4 * 512), 256, 0, stream>>>(w2, wp2, 32, 32, 64, 27, 4);
    pack_weights_kernel<<<nb(54 * 2 * 512), 256, 0, stream>>>(w3, wp3, 64, 64, 32, 54, 2);
    pack_weights_kernel<<<nb(27 * 1 * 512), 256, 0, stream>>>(w4, wp4, 32, 32, 3, 27, 1);

    const int levels[2] = {48, 96};
    for (int lvl = 0; lvl < 2; ++lvl) {
        const int Dl = levels[lvl];
        const int Nl = Dl * Dl * Dl;
        const int Npl = (Dl + 2) * (Dl + 2) * (Dl + 2);

        resize_tri_kernel<<<nb(Nl), 256, 0, stream>>>(image, si, 1, DF, DF, DF, Dl, Dl, Dl, 1.f);
        resize_tri_kernel<<<nb(Nl), 256, 0, stream>>>(moving, smov, 1, DF, DF, DF, Dl, Dl, Dl, 1.f);

        // zero padded activation buffers once per level (halo stays zero; the
        // interior is fully rewritten before every conv that reads it)
        fill_zero_kernel<<<nb(Npl * 16 / 2), 256, 0, stream>>>((float*)act16, Npl * 16 / 2);
        fill_zero_kernel<<<nb(Npl * 32 / 2), 256, 0, stream>>>((float*)act32, Npl * 32 / 2);
        fill_zero_kernel<<<nb(Npl * 64 / 2), 256, 0, stream>>>((float*)act64, Npl * 64 / 2);

        if (lvl == 0) {
            fill_zero_kernel<<<nb(3 * Nl), 256, 0, stream>>>(vf, 3 * Nl);
        } else {
            // match_vf: upsample 48^3 -> 96^3 and scale displacements by 2
            resize_tri_kernel<<<nb(3 * Nl), 256, 0, stream>>>(vf, vf2, 3, 48, 48, 48, Dl, Dl, Dl, 2.0f);
            scale_copy_kernel<<<nb(3 * Nl), 256, 0, stream>>>(vf2, vf, 1.f, 3 * Nl);
        }

        for (int it = 0; it < 2; ++it) {
            warp_kernel<<<nb(Nl), 256, 0, stream>>>(smov, vf, warped, Dl, Dl, Dl);
            feat_kernel<<<nb(Nl), 256, 0, stream>>>(warped, si, act16, Dl, Dl, Dl);

            // conv1: 5(->16 pad)->32, K=432 padded to 448 (14 K-tiles), 2 N-tiles
            if (Dl == 48) conv3d_wmma_kernel<2, 16, 14, 48><<<Nl / 64, 128, 0, stream>>>(act16, wp1, conv, 32);
            else          conv3d_wmma_kernel<2, 16, 14, 96><<<Nl / 64, 128, 0, stream>>>(act16, wp1, conv, 32);
            inorm_stats_kernel<<<32, 256, 0, stream>>>(conv, stats, Nl);
            norm_mish_nhwc_kernel<32><<<nb(Nl), 256, 0, stream>>>(conv, stats, act32, Dl, Dl, Dl);

            // conv2: 32->64, K=864 (27 K-tiles), 4 N-tiles
            if (Dl == 48) conv3d_wmma_kernel<4, 32, 27, 48><<<Nl / 64, 128, 0, stream>>>(act32, wp2, conv, 64);
            else          conv3d_wmma_kernel<4, 32, 27, 96><<<Nl / 64, 128, 0, stream>>>(act32, wp2, conv, 64);
            inorm_stats_kernel<<<64, 256, 0, stream>>>(conv, stats, Nl);
            norm_mish_nhwc_kernel<64><<<nb(Nl), 256, 0, stream>>>(conv, stats, act64, Dl, Dl, Dl);

            // conv3: 64->32, K=1728 (54 K-tiles), 2 N-tiles
            if (Dl == 48) conv3d_wmma_kernel<2, 64, 54, 48><<<Nl / 64, 128, 0, stream>>>(act64, wp3, conv, 32);
            else          conv3d_wmma_kernel<2, 64, 54, 96><<<Nl / 64, 128, 0, stream>>>(act64, wp3, conv, 32);
            inorm_stats_kernel<<<32, 256, 0, stream>>>(conv, stats, Nl);
            norm_mish_nhwc_kernel<32><<<nb(Nl), 256, 0, stream>>>(conv, stats, act32, Dl, Dl, Dl);

            // conv4: 32->3 (+b4 in vf update), K=864 (27 K-tiles), 1 N-tile
            if (Dl == 48) conv3d_wmma_kernel<1, 32, 27, 48><<<Nl / 64, 128, 0, stream>>>(act32, wp4, conv, 3);
            else          conv3d_wmma_kernel<1, 32, 27, 96><<<Nl / 64, 128, 0, stream>>>(act32, wp4, conv, 3);

            vf_update_kernel<<<nb(3 * Nl), 256, 0, stream>>>(vf, conv, b4, Nl);

            smooth_axis_kernel<<<nb(3 * Nl), 256, 0, stream>>>(vf, vf2, 3, Dl, Dl, Dl, 0);
            smooth_axis_kernel<<<nb(3 * Nl), 256, 0, stream>>>(vf2, vf, 3, Dl, Dl, Dl, 1);
            smooth_axis_kernel<<<nb(3 * Nl), 256, 0, stream>>>(vf, vf2, 3, Dl, Dl, Dl, 2);
            scale_copy_kernel<<<nb(3 * Nl), 256, 0, stream>>>(vf2, vf, 1.f, 3 * Nl);
        }
    }

    // final match_vf is identity (already full-res); warp moving + emit outputs
    float* outp = (float*)d_out;
    warp_kernel<<<nb(NF), 256, 0, stream>>>(moving, vf, outp, DF, DF, DF);
    scale_copy_kernel<<<nb(3 * NF), 256, 0, stream>>>(vf, outp + NF, 1.f, 3 * NF);
}
